// RWKV_TimeMix_20461224198645
// MI455X (gfx1250) — matrix operand order repaired and checked by reading, not compiled, not run on hardware
//
#include <hip/hip_runtime.h>
#include <hip/hip_bf16.h>

// Problem constants (match reference: B=4, T=4096, D=1024)
constexpr int Bsz = 4;
constexpr int T   = 4096;
constexpr int D   = 1024;
constexpr size_t NE = (size_t)Bsz * T * D;   // 16,777,216 elements
constexpr int M_TOTAL = Bsz * T;             // 16384 GEMM rows
constexpr unsigned WMAT_BYTES = (unsigned)D * (unsigned)D * 2u;  // 2 MiB per bf16 weight

// LDS staging: rows of 32 K bf16 padded 32->40 elems (80 B) for bank spread
constexpr int LROW      = 80;                // bytes per padded row
constexpr int RK_MATSZ  = 32 * LROW;         // 2560 B  (rkv kernel: 32 N-rows/mat)
constexpr int RK_STAGE  = 3 * RK_MATSZ;      // 7680 B
constexpr int OUT_MATSZ = 64 * LROW;         // 5120 B  (out kernel: 64 N-rows)

typedef unsigned short u16;
typedef __attribute__((ext_vector_type(16))) __bf16 v16bf;
typedef __attribute__((ext_vector_type(8)))  float  v8f;

// ---------- scalar fp32 <-> bf16 helpers (round-to-nearest-even) ----------
__device__ __forceinline__ u16 f2bf(float f) {
    unsigned int u = __float_as_uint(f);
    unsigned int r = u + 0x7FFFu + ((u >> 16) & 1u);
    return (u16)(r >> 16);
}
__device__ __forceinline__ float bf2f(u16 h) {
    return __uint_as_float(((unsigned int)h) << 16);
}

__device__ __forceinline__ v8f wmma_bf16(v16bf a, v16bf b, v8f c) {
    return __builtin_amdgcn_wmma_f32_16x16x32_bf16(false, a, false, b,
                                                   (short)0, c, false, false);
}

union FragU { uint4 q[2]; v16bf v; };

// ---------- CDNA5 async global->LDS copy (ASYNCcnt-tracked, no VGPR staging) ----------
typedef __attribute__((address_space(3))) unsigned char lds_byte;

// LDS dest address comes from the vdst VGPR; global address = SGPR base + 32-bit VGPR offset.
// base MUST be wave-uniform (SGPR pair). No memory clobber: the written LDS buffer
// is only read after wait_async0() + __syncthreads(), which carry the fences.
__device__ __forceinline__ void async_b128_to_lds(void* ldsPtr,
                                                  const u16* base,
                                                  unsigned byteOff) {
    unsigned lds = (unsigned)(size_t)(lds_byte*)ldsPtr;   // 32-bit LDS byte address
    asm volatile("global_load_async_to_lds_b128 %0, %1, %2"
                 :: "v"(lds), "v"(byteOff),
                    "s"((unsigned long long)(size_t)base));
}
__device__ __forceinline__ void wait_async0() {
    asm volatile("s_wait_asynccnt 0" ::: "memory");
}

// A matrix 16x32 (MxK), contiguous k order (same as B fragment), direct from global:
//   lane m (0..15): K=+0..15; lane m+16: K=+16..31
__device__ __forceinline__ v16bf load_fragA(const u16* __restrict__ base, unsigned off) {
    FragU u;
    const uint4* p = reinterpret_cast<const uint4*>(base + off);
    u.q[0] = p[0];
    u.q[1] = p[1];
    return u.v;
}
// B matrix 32x16 (KxN) fragment from LDS (padded rows):
//   lane n (0..15): K=+0..15 of column n; lane n+16: K=+16..31
__device__ __forceinline__ v16bf load_fragB_lds(const char* buf, unsigned byteOff) {
    FragU u;
    const uint4* p = reinterpret_cast<const uint4*>(buf + byteOff);
    u.q[0] = p[0];
    u.q[1] = p[1];
    return u.v;
}

// ---------- fp32 -> bf16 convert (vectorized x4) ----------
__global__ void convert_f2bf_kernel(const float4* __restrict__ src,
                                    ushort4* __restrict__ dst, int n4) {
    int i = blockIdx.x * blockDim.x + threadIdx.x;
    if (i < n4) {
        float4 f = src[i];
        ushort4 o;
        o.x = f2bf(f.x); o.y = f2bf(f.y); o.z = f2bf(f.z); o.w = f2bf(f.w);
        dst[i] = o;
    }
}

// ---------- fused r/k/v GEMM: r=sigmoid(xWr^T) (bf16), kv=(xWk^T)*(xWv^T) (f32) ----------
// grid: (D/32, M_TOTAL/128), block 256 (8 waves); wave: 16(M) x 32(N); K-step 32.
// wbase points at Wr_bf16; Wk/Wv follow contiguously at +1/+2 * WMAT_BYTES, so all
// async copies use ONE uniform SGPR base with 32-bit offsets.
__global__ void gemm_rkv_kernel(const u16* __restrict__ xbf,
                                const u16* __restrict__ wbase,
                                u16* __restrict__ r_out,
                                float* __restrict__ kv_out) {
    __shared__ char smem[2 * RK_STAGE];

    const int tid  = threadIdx.x;
    const int wave = tid >> 5;
    const int lane = tid & 31;
    const int m0 = blockIdx.y * 128 + wave * 16;
    const int n0 = blockIdx.x * 32;

    // cooperative B staging: 3 mats x 32 rows x 4 chunks = 384 16B-chunks/stage
    //   chunk #tid      : mat = tid>>7 (Wr/Wk), row = (tid&127)>>2, kc = tid&3
    //   chunk #(256+tid): mat = 2 (Wv), tid<128 only, row = tid>>2
    const int row1 = (tid & 127) >> 2;
    const int kc   = tid & 3;
    unsigned gb1 = (unsigned)(tid >> 7) * WMAT_BYTES
                 + (((unsigned)(n0 + row1) << 10) + (unsigned)kc * 8u) * 2u;  // bytes
    const unsigned soff1 = (unsigned)(tid >> 7) * RK_MATSZ
                         + (unsigned)row1 * LROW + (unsigned)kc * 16u;
    const int row2 = tid >> 2;                     // valid when tid < 128
    unsigned gb2 = 2u * WMAT_BYTES
                 + (((unsigned)(n0 + row2) << 10) + (unsigned)kc * 8u) * 2u;
    const unsigned soff2 = 2u * RK_MATSZ + (unsigned)row2 * LROW + (unsigned)kc * 16u;

    unsigned offA = ((unsigned)(m0 + (lane & 15)) << 10) + ((unsigned)(lane >> 4) << 4);

    const unsigned fOff0 = (unsigned)(lane & 15) * LROW + ((unsigned)(lane >> 4)) * 32u;
    const unsigned fOff1 = fOff0 + 16u * LROW;

    v8f accR[2] = {}, accK[2] = {}, accV[2] = {};

    // ---- preload stage 0 (async straight into LDS) ----
    async_b128_to_lds(smem + soff1, wbase, gb1);
    if (tid < 128) async_b128_to_lds(smem + soff2, wbase, gb2);
    gb1 += 64; gb2 += 64;
    v16bf aCur = load_fragA(xbf, offA);
    offA += 32;
    wait_async0();
    __syncthreads();

#pragma unroll
    for (int kstep = 0; kstep < D / 32; ++kstep) {
        const int cur = kstep & 1;
        const char* bufC = smem + cur * RK_STAGE;
        char* bufN = smem + (cur ^ 1) * RK_STAGE;
        const bool more = (kstep + 1) < (D / 32);

        v16bf aNxt;
        if (more) {                          // async prefetch into bufN, above WMMA chain
            async_b128_to_lds(bufN + soff1, wbase, gb1);
            if (tid < 128) async_b128_to_lds(bufN + soff2, wbase, gb2);
            gb1 += 64; gb2 += 64;
            aNxt = load_fragA(xbf, offA);
            offA += 32;
        }

        // depth-2 fragment pipeline over 6 WMMAs: (R,K,V) x (nt=0,1)
        v16bf b0 = load_fragB_lds(bufC + 0 * RK_MATSZ, fOff0);
        v16bf b1 = load_fragB_lds(bufC + 1 * RK_MATSZ, fOff0);
        accR[0] = wmma_bf16(aCur, b0, accR[0]);
        b0 = load_fragB_lds(bufC + 2 * RK_MATSZ, fOff0);
        accK[0] = wmma_bf16(aCur, b1, accK[0]);
        b1 = load_fragB_lds(bufC + 0 * RK_MATSZ, fOff1);
        accV[0] = wmma_bf16(aCur, b0, accV[0]);
        b0 = load_fragB_lds(bufC + 1 * RK_MATSZ, fOff1);
        accR[1] = wmma_bf16(aCur, b1, accR[1]);
        b1 = load_fragB_lds(bufC + 2 * RK_MATSZ, fOff1);
        accK[1] = wmma_bf16(aCur, b0, accK[1]);
        accV[1] = wmma_bf16(aCur, b1, accV[1]);

        if (more) aCur = aNxt;
        wait_async0();                       // my bufN writes landed in LDS
        __syncthreads();                     // everyone's landed
    }

    // C/D layout: VGPR j -> M = j (lanes 0-15, N=lane) or M = j+8 (lanes 16-31)
    const int nl = lane & 15;
    const int mb = (lane >> 4) * 8;
#pragma unroll
    for (int nt = 0; nt < 2; ++nt) {
#pragma unroll
        for (int j = 0; j < 8; ++j) {
            size_t o = (size_t)(m0 + mb + j) * D + (n0 + nt * 16 + nl);
            float rv = 1.0f / (1.0f + __expf(-accR[nt][j]));
            r_out[o]  = f2bf(rv);
            kv_out[o] = accK[nt][j] * accV[nt][j];
        }
    }
}

// ---------- decayed scan along T, mirrors reference clip/exp semantics ----------
__global__ void scan_kernel(float* __restrict__ kv_rs,
                            const u16* __restrict__ rbf,
                            const float* __restrict__ decay) {
    int idx = blockIdx.x * blockDim.x + threadIdx.x;   // 0..B*D-1
    int b = idx >> 10;
    int d = idx & (D - 1);
    float dec = 1.0f / (1.0f + __expf(-decay[d]));
    float ld  = __logf(fmaxf(dec, 1e-7f));
    float* kvp = kv_rs + (size_t)b * T * D + d;
    const u16* rp = rbf + (size_t)b * T * D + d;
    float cum = 0.0f;
    for (int t = 0; t < T; ++t) {
        float scale = __expf((float)t * ld);           // underflows to 0 like reference
        cum += kvp[(size_t)t * D] / fmaxf(scale, 1e-10f);
        float state = cum * scale;
        float r = bf2f(rp[(size_t)t * D]);
        kvp[(size_t)t * D] = r * state;
    }
}

// ---------- RMSNorm over D, output bf16 ----------
__global__ void rmsnorm_kernel(const float* __restrict__ rs,
                               const float* __restrict__ ln_w,
                               u16* __restrict__ ybf) {
    const int row = blockIdx.x;                        // 0..M_TOTAL-1
    const float* p = rs + (size_t)row * D;
    float v[4];
    float ss = 0.0f;
#pragma unroll
    for (int i = 0; i < 4; ++i) {
        v[i] = p[threadIdx.x + i * 256];
        ss += v[i] * v[i];
    }
#pragma unroll
    for (int off = 16; off > 0; off >>= 1)
        ss += __shfl_xor(ss, off, 32);
    __shared__ float red[8];
    if ((threadIdx.x & 31) == 0) red[threadIdx.x >> 5] = ss;
    __syncthreads();
    float tot = 0.0f;
#pragma unroll
    for (int w = 0; w < 8; ++w) tot += red[w];
    float inv = rsqrtf(tot * (1.0f / (float)D) + 1e-6f);
#pragma unroll
    for (int i = 0; i < 4; ++i) {
        int c = threadIdx.x + i * 256;
        ybf[(size_t)row * D + c] = f2bf(v[i] * inv * ln_w[c]);
    }
}

// ---------- output GEMM: out = y @ Wo^T (f32), async-LDS pipeline, 4 N-tiles ----------
// grid: (D/64, M_TOTAL/128), block 256 (8 waves); wave: 16(M) x 64(N)
__global__ void gemm_out_kernel(const u16* __restrict__ ybf,
                                const u16* __restrict__ wo,
                                float* __restrict__ out) {
    __shared__ char smem[2 * OUT_MATSZ];

    const int tid  = threadIdx.x;
    const int wave = tid >> 5;
    const int lane = tid & 31;
    const int m0 = blockIdx.y * 128 + wave * 16;
    const int n0 = blockIdx.x * 64;

    const int crow = tid >> 2;                     // 0..63
    const int ckc  = tid & 3;
    unsigned gbB = (((unsigned)(n0 + crow) << 10) + (unsigned)ckc * 8u) * 2u;  // bytes
    const unsigned soffB = (unsigned)crow * LROW + (unsigned)ckc * 16u;

    unsigned offA = ((unsigned)(m0 + (lane & 15)) << 10) + ((unsigned)(lane >> 4) << 4);
    const unsigned fOffB = (unsigned)(lane & 15) * LROW + ((unsigned)(lane >> 4)) * 32u;

    v8f acc[4] = {};

    async_b128_to_lds(smem + soffB, wo, gbB);
    gbB += 64;
    v16bf aCur = load_fragA(ybf, offA);
    offA += 32;
    wait_async0();
    __syncthreads();

#pragma unroll
    for (int kstep = 0; kstep < D / 32; ++kstep) {
        const int cur = kstep & 1;
        const char* bufC = smem + cur * OUT_MATSZ;
        char* bufN = smem + (cur ^ 1) * OUT_MATSZ;
        const bool more = (kstep + 1) < (D / 32);

        v16bf aNxt;
        if (more) {
            async_b128_to_lds(bufN + soffB, wo, gbB);
            gbB += 64;
            aNxt = load_fragA(ybf, offA);
            offA += 32;
        }

        // depth-2 fragment pipeline over 4 WMMAs
        v16bf b0 = load_fragB_lds(bufC, fOffB);
        v16bf b1 = load_fragB_lds(bufC, fOffB + 16u * LROW);
        acc[0] = wmma_bf16(aCur, b0, acc[0]);
        b0 = load_fragB_lds(bufC, fOffB + 32u * LROW);
        acc[1] = wmma_bf16(aCur, b1, acc[1]);
        b1 = load_fragB_lds(bufC, fOffB + 48u * LROW);
        acc[2] = wmma_bf16(aCur, b0, acc[2]);
        acc[3] = wmma_bf16(aCur, b1, acc[3]);

        if (more) aCur = aNxt;
        wait_async0();
        __syncthreads();
    }

    const int nl = lane & 15;
    const int mb = (lane >> 4) * 8;
#pragma unroll
    for (int nt = 0; nt < 4; ++nt) {
#pragma unroll
        for (int j = 0; j < 8; ++j) {
            out[(size_t)(m0 + mb + j) * D + (n0 + nt * 16 + nl)] = acc[nt][j];
        }
    }
}

extern "C" void kernel_launch(void* const* d_in, const int* in_sizes, int n_in,
                              void* d_out, int out_size, void* d_ws, size_t ws_size,
                              hipStream_t stream) {
    const float* x     = (const float*)d_in[0];
    const float* Wr    = (const float*)d_in[1];
    const float* Wk    = (const float*)d_in[2];
    const float* Wv    = (const float*)d_in[3];
    const float* Wo    = (const float*)d_in[4];
    const float* decay = (const float*)d_in[5];
    const float* ln_w  = (const float*)d_in[6];
    float* out = (float*)d_out;

    // workspace layout: xbf 32M | Wr|Wk|Wv|Wo bf16 (contiguous, 8M) | rbf 32M |
    //                   kv/rs 64M | ybf 32M  (~168 MB)
    char* ws = (char*)d_ws;
    u16* xbf  = (u16*)ws;
    u16* wrbf = xbf + NE;                       // Wk = wrbf + D*D, Wv = wrbf + 2*D*D
    u16* wkbf = wrbf + (size_t)D * D;
    u16* wvbf = wkbf + (size_t)D * D;
    u16* wobf = wvbf + (size_t)D * D;
    u16* rbf  = wobf + (size_t)D * D;
    float* kv = (float*)(rbf + NE);
    u16* ybf  = (u16*)(kv + NE);

    // 1) fp32 -> bf16 conversions (vectorized x4)
    {
        int n4 = (int)(NE / 4);
        convert_f2bf_kernel<<<(n4 + 255) / 256, 256, 0, stream>>>(
            (const float4*)x, (ushort4*)xbf, n4);
        int nw4 = (D * D) / 4;
        convert_f2bf_kernel<<<(nw4 + 255) / 256, 256, 0, stream>>>(
            (const float4*)Wr, (ushort4*)wrbf, nw4);
        convert_f2bf_kernel<<<(nw4 + 255) / 256, 256, 0, stream>>>(
            (const float4*)Wk, (ushort4*)wkbf, nw4);
        convert_f2bf_kernel<<<(nw4 + 255) / 256, 256, 0, stream>>>(
            (const float4*)Wv, (ushort4*)wvbf, nw4);
        convert_f2bf_kernel<<<(nw4 + 255) / 256, 256, 0, stream>>>(
            (const float4*)Wo, (ushort4*)wobf, nw4);
    }

    // 2) fused r/k/v GEMM (WMMA bf16 -> f32, async-LDS double-buffered B)
    gemm_rkv_kernel<<<dim3(D / 32, M_TOTAL / 128), 256, 0, stream>>>(
        xbf, wrbf, rbf, kv);

    // 3) decayed scan along T (overwrites kv with r*state)
    scan_kernel<<<(Bsz * D) / 256, 256, 0, stream>>>(kv, rbf, decay);

    // 4) RMSNorm over D -> bf16 y
    rmsnorm_kernel<<<M_TOTAL, 256, 0, stream>>>(kv, ln_w, ybf);

    // 5) output GEMM (WMMA bf16 -> f32, async-LDS double-buffered B)
    gemm_out_kernel<<<dim3(D / 64, M_TOTAL / 128), 256, 0, stream>>>(ybf, wobf, out);
}
